// TransformerEncoderLayer_51256139710914
// MI455X (gfx1250) — matrix operand order repaired and checked by reading, not compiled, not run on hardware
//
#include <hip/hip_runtime.h>

// ---------------- problem constants ----------------
#define D_MODEL 1024
#define NHEADS  16
#define HDIM    64
#define SEQ     2048
#define BATCH   4
#define MTOT    (BATCH*SEQ)     // 8192 token rows
#define EPS     1e-5f

typedef __bf16 bf16;
typedef __attribute__((ext_vector_type(16))) __bf16 v16bf;
typedef __attribute__((ext_vector_type(8)))  __bf16 v8bf;
typedef __attribute__((ext_vector_type(8)))  float  v8f;

union Frag16 { v16bf v; v8bf h[2]; };

__device__ inline v8f wmma_bf16(v16bf a, v16bf b, v8f c) {
  // D(f32 16x16) = A(bf16 16x32) * B(bf16 32x16) + C
  return __builtin_amdgcn_wmma_f32_16x16x32_bf16(false, a, false, b, (short)0, c,
                                                 false, false);
}

// gfx1250 async global->LDS copy (ASYNCcnt-tracked). lds_off is the byte
// offset inside the static group segment (segment base == 0).
__device__ inline void async_g2l_b128(unsigned lds_off, const void* gaddr) {
  asm volatile("global_load_async_to_lds_b128 %0, %1, off"
               :: "v"(lds_off), "v"(gaddr) : "memory");
}
__device__ inline void wait_asynccnt0() {
  asm volatile("s_wait_asynccnt 0x0" ::: "memory");
}
__device__ inline void wait_asynccnt1() {
  asm volatile("s_wait_asynccnt 0x1" ::: "memory");
}

// ---------------- fp32 -> bf16 weight conversion ----------------
__global__ void __launch_bounds__(256)
k_w2bf(const float* __restrict__ w0, const float* __restrict__ w1,
       const float* __restrict__ w2, const float* __restrict__ w3,
       const float* __restrict__ w4, const float* __restrict__ w5,
       bf16* __restrict__ out) {
  const float* srcs[6] = {w0, w1, w2, w3, w4, w5};
  const int w = blockIdx.y;
  const int i = blockIdx.x * blockDim.x + threadIdx.x;
  out[(size_t)w * (D_MODEL * D_MODEL) + i] = (bf16)srcs[w][i];
}

// ---------------- layernorm (fp32 in -> bf16 out) ----------------
__global__ void __launch_bounds__(256)
k_layernorm(const float* __restrict__ x, const float* __restrict__ g,
            const float* __restrict__ be, bf16* __restrict__ y) {
  __shared__ float red[256];
  const int row = blockIdx.x;
  const int t = threadIdx.x;
  const float* xr = x + (size_t)row * D_MODEL;
  float vals[4];
  float s = 0.f, s2 = 0.f;
#pragma unroll
  for (int k = 0; k < 4; ++k) {
    const float v = xr[t + k * 256];
    vals[k] = v; s += v; s2 += v * v;
  }
  red[t] = s; __syncthreads();
  for (int st = 128; st > 0; st >>= 1) { if (t < st) red[t] += red[t + st]; __syncthreads(); }
  const float mu = red[0] * (1.f / D_MODEL);
  __syncthreads();
  red[t] = s2; __syncthreads();
  for (int st = 128; st > 0; st >>= 1) { if (t < st) red[t] += red[t + st]; __syncthreads(); }
  const float var = red[0] * (1.f / D_MODEL) - mu * mu;
  const float rstd = rsqrtf(var + EPS);
  bf16* yr = y + (size_t)row * D_MODEL;
#pragma unroll
  for (int k = 0; k < 4; ++k) {
    const int c = t + k * 256;
    yr[c] = (bf16)((vals[k] - mu) * rstd * g[c] + be[c]);
  }
}

// ---------------- generic bf16 WMMA GEMM:  out = act(A @ W^T + bias) [+resid] ----
// A: [M,K] bf16 row-major.  W: [N,K] bf16 row-major.
// flags: 1=ReLU  2=add fp32 residual  4=bf16 output  8=head-reorder bf16 output
// Depth-2 software pipeline with COMPILE-TIME ping-pong slots (a0/a1, b0/b1):
// two K-steps per iteration so every local is constant-indexed (SROA -> VGPRs).
__global__ void __launch_bounds__(256)
k_gemm(const bf16* __restrict__ A, const bf16* __restrict__ W,
       const float* __restrict__ bias, const float* __restrict__ resid,
       float* __restrict__ outF, bf16* __restrict__ outB,
       int M, int N, int K, int flags) {
  const int tid  = threadIdx.x;
  const int lane = tid & 31;
  const int wave = tid >> 5;            // 8 waves: 2 (M) x 4 (N)
  const int lh   = lane >> 4;           // lane half
  const int ln   = lane & 15;
  const int m0 = blockIdx.x * 32  + (wave & 1) * 16;
  const int n0 = blockIdx.y * 256 + (wave >> 1) * 64;

  v8f acc[4] = {};
  const bf16* arow = A + (size_t)(m0 + ln) * K;
  const bf16* wrow[4];
#pragma unroll
  for (int t = 0; t < 4; ++t)
    wrow[t] = W + (size_t)(n0 + t * 16 + ln) * K + lh * 16;

  Frag16 a0, a1;
  v16bf b0[4], b1[4];
  a0.h[0] = *(const v8bf*)(arow +     lh * 16);
  a0.h[1] = *(const v8bf*)(arow + 8 + lh * 16);
  a1.h[0] = *(const v8bf*)(arow + 32 +     lh * 16);
  a1.h[1] = *(const v8bf*)(arow + 32 + 8 + lh * 16);
#pragma unroll
  for (int t = 0; t < 4; ++t) b0[t] = *(const v16bf*)(wrow[t]);
#pragma unroll
  for (int t = 0; t < 4; ++t) b1[t] = *(const v16bf*)(wrow[t] + 32);

  for (int kb = 0; kb < K; kb += 64) {
    const int k2 = (kb + 64 < K) ? kb + 64 : kb;   // refill targets (tail: reload)
    const int k3 = (kb + 96 < K) ? kb + 96 : kb;
    __builtin_prefetch(arow + kb + 192, 0, 1);
    // consume slot 0, refill it for kb+64
#pragma unroll
    for (int t = 0; t < 4; ++t) acc[t] = wmma_bf16(a0.v, b0[t], acc[t]);
    a0.h[0] = *(const v8bf*)(arow + k2 +     lh * 16);
    a0.h[1] = *(const v8bf*)(arow + k2 + 8 + lh * 16);
#pragma unroll
    for (int t = 0; t < 4; ++t) b0[t] = *(const v16bf*)(wrow[t] + k2);
    // consume slot 1, refill it for kb+96
#pragma unroll
    for (int t = 0; t < 4; ++t) acc[t] = wmma_bf16(a1.v, b1[t], acc[t]);
    a1.h[0] = *(const v8bf*)(arow + k3 +     lh * 16);
    a1.h[1] = *(const v8bf*)(arow + k3 + 8 + lh * 16);
#pragma unroll
    for (int t = 0; t < 4; ++t) b1[t] = *(const v16bf*)(wrow[t] + k3);
  }

#pragma unroll
  for (int t = 0; t < 4; ++t) {
    const int n = n0 + t * 16 + ln;
    const float bv = bias ? bias[n] : 0.f;
#pragma unroll
    for (int i = 0; i < 8; ++i) {
      const int m = m0 + i + lh * 8;    // C layout: VGPR i -> row i (+8 upper half)
      float v = acc[t][i] + bv;
      if (flags & 1) v = fmaxf(v, 0.f);
      if (flags & 2) v += resid[(size_t)m * N + n];
      if (flags & 4) {
        size_t idx;
        if (flags & 8) {                // token-major -> [B,H,S,HDIM]
          const int b_ = m / SEQ, s_ = m % SEQ;
          const int h_ = n / HDIM, hd = n % HDIM;
          idx = (((size_t)b_ * NHEADS + h_) * SEQ + s_) * HDIM + hd;
        } else idx = (size_t)m * N + n;
        outB[idx] = (bf16)v;
      } else {
        outF[(size_t)m * N + n] = v;
      }
    }
  }
}

// ---------------- flash attention: one block per (b,h,128 q-rows) --------------
// LDS arena (single array so byte offsets == hardware LDS addresses):
//   [0,     4096)  K ping  : 32 keys x 64 hd bf16 (async)
//   [4096,  8192)  K pong
//   [8192, 12288)  Vt ping : 64 hd x 32 keys bf16 (transposed)
//   [12288,16384)  Vt pong
//   [16384,24576)  P       : 8 waves x 16x32 probs
#define SK0_OFF  0
#define SK1_OFF  4096
#define SV0_OFF  8192
#define SV1_OFF  12288
#define SP_OFF   16384
__global__ void __launch_bounds__(256)
k_attn(const bf16* __restrict__ Q, const bf16* __restrict__ Kg,
       const bf16* __restrict__ Vg, bf16* __restrict__ AO) {
  __shared__ __align__(32) bf16 smem[(SP_OFF + 8 * 16 * 32 * 2) / 2];

  const int tid  = threadIdx.x;
  const int lane = tid & 31;
  const int wave = tid >> 5;
  const int lh   = lane >> 4;
  const int ln   = lane & 15;
  const int b = blockIdx.z, h = blockIdx.y;
  const int q0 = blockIdx.x * 128 + wave * 16;
  const size_t headOff = ((size_t)b * NHEADS + h) * SEQ * HDIM;

  // Q fragments (16 rows x 64 hd = 2 K-steps)
  Frag16 qa0, qa1;
  const bf16* qrow = Q + headOff + (size_t)(q0 + ln) * HDIM;
  qa0.h[0] = *(const v8bf*)(qrow +     lh * 16);
  qa0.h[1] = *(const v8bf*)(qrow + 8 + lh * 16);
  qa1.h[0] = *(const v8bf*)(qrow + 32 +     lh * 16);
  qa1.h[1] = *(const v8bf*)(qrow + 32 + 8 + lh * 16);

  v8f o[4] = {};
  float mrow[8], lrow[8];
#pragma unroll
  for (int i = 0; i < 8; ++i) { mrow[i] = -3.0e38f; lrow[i] = 0.f; }

  const int key = tid >> 3;             // 0..31 (staging role)
  const int hd8 = (tid & 7) * 8;
  const bf16* kbase = Kg + headOff;
  const bf16* vbase = Vg + headOff;
  const unsigned kOffLane = (unsigned)((key * HDIM + hd8) * 2);

  const int NC = SEQ / 32;
  // prologue: chunk 0 in flight / staged
  async_g2l_b128(SK0_OFF + kOffLane, kbase + (size_t)key * HDIM + hd8);
  {
    const v8bf vv = *(const v8bf*)(vbase + (size_t)key * HDIM + hd8);
    bf16* sV0 = smem + SV0_OFF / 2;
#pragma unroll
    for (int j = 0; j < 8; ++j) sV0[(hd8 + j) * 32 + key] = vv[j];
  }

  for (int c = 0; c < NC; ++c) {
    const int buf = c & 1;
    const bool more = (c + 1) < NC;
    // runtime-selected integer offsets (no constant generic-pointer arrays)
    bf16* sKb = smem + (buf ? (SK1_OFF / 2) : (SK0_OFF / 2));
    bf16* sVb = smem + (buf ? (SV1_OFF / 2) : (SV0_OFF / 2));
    bf16* sVn = smem + (buf ? (SV0_OFF / 2) : (SV1_OFF / 2));
    v8bf vvn = {};
    if (more) {       // overlap next chunk's DMA + V fetch with this chunk's math
      const unsigned ldsKn = (buf ? SK0_OFF : SK1_OFF) + kOffLane;
      async_g2l_b128(ldsKn, kbase + (size_t)((c + 1) * 32 + key) * HDIM + hd8);
      vvn = *(const v8bf*)(vbase + (size_t)((c + 1) * 32 + key) * HDIM + hd8);
    }
    if (more) wait_asynccnt1(); else wait_asynccnt0();  // oldest async (buf) landed
    __syncthreads();                                    // all lanes' K/Vt[buf] visible

    // scores: two 16x16 tiles covering 32 keys
    v8f s0 = {}, s1 = {};
    {
      const v16bf bk00 = *(const v16bf*)(sKb + (ln)      * HDIM +      lh * 16);
      const v16bf bk10 = *(const v16bf*)(sKb + (16 + ln) * HDIM +      lh * 16);
      const v16bf bk01 = *(const v16bf*)(sKb + (ln)      * HDIM + 32 + lh * 16);
      const v16bf bk11 = *(const v16bf*)(sKb + (16 + ln) * HDIM + 32 + lh * 16);
      s0 = wmma_bf16(qa0.v, bk00, s0);
      s1 = wmma_bf16(qa0.v, bk10, s1);
      s0 = wmma_bf16(qa1.v, bk01, s0);
      s1 = wmma_bf16(qa1.v, bk11, s1);
    }
    const float scale = 0.125f;          // 1/sqrt(64)
#pragma unroll
    for (int i = 0; i < 8; ++i) { s0[i] *= scale; s1[i] *= scale; }

    // online softmax (row stats replicated across the 16-lane half)
#pragma unroll
    for (int i = 0; i < 8; ++i) {
      float r = fmaxf(s0[i], s1[i]);
      r = fmaxf(r, __shfl_xor(r, 1, 32));
      r = fmaxf(r, __shfl_xor(r, 2, 32));
      r = fmaxf(r, __shfl_xor(r, 4, 32));
      r = fmaxf(r, __shfl_xor(r, 8, 32));
      const float mnew = fmaxf(mrow[i], r);
      const float alpha = __expf(mrow[i] - mnew);
      mrow[i] = mnew; lrow[i] *= alpha;
      o[0][i] *= alpha; o[1][i] *= alpha; o[2][i] *= alpha; o[3][i] *= alpha;
      const float p0 = __expf(s0[i] - mnew);
      const float p1 = __expf(s1[i] - mnew);
      s0[i] = p0; s1[i] = p1;
      float ps = p0 + p1;
      ps += __shfl_xor(ps, 1, 32);
      ps += __shfl_xor(ps, 2, 32);
      ps += __shfl_xor(ps, 4, 32);
      ps += __shfl_xor(ps, 8, 32);
      lrow[i] += ps;
    }

    // relayout C-tile probabilities -> A-fragment via per-wave LDS scratch.
    // No barrier needed: LDS ops from one wave complete in order.
    bf16* P = smem + SP_OFF / 2 + wave * (16 * 32);
#pragma unroll
    for (int i = 0; i < 8; ++i) {
      const int r = i + lh * 8;
      P[r * 32 + ln]      = (bf16)s0[i];
      P[r * 32 + 16 + ln] = (bf16)s1[i];
    }
    Frag16 pf;
    pf.h[0] = *(const v8bf*)(P + ln * 32 +     lh * 16);
    pf.h[1] = *(const v8bf*)(P + ln * 32 + 8 + lh * 16);
#pragma unroll
    for (int t = 0; t < 4; ++t) {
      const v16bf vb = *(const v16bf*)(sVb + (t * 16 + ln) * 32 + lh * 16);
      o[t] = wmma_bf16(pf.v, vb, o[t]);
    }

    // stage next chunk's V (other buffer); its readers sync at next barrier
    if (more) {
#pragma unroll
      for (int j = 0; j < 8; ++j) sVn[(hd8 + j) * 32 + key] = vvn[j];
    }
  }

  // normalize and store token-major bf16 [B,S,D]
#pragma unroll
  for (int i = 0; i < 8; ++i) {
    const float inv = 1.f / lrow[i];
    const int m = q0 + i + lh * 8;
    const size_t tok = (size_t)b * SEQ + m;
#pragma unroll
    for (int t = 0; t < 4; ++t)
      AO[tok * D_MODEL + h * HDIM + t * 16 + ln] = (bf16)(o[t][i] * inv);
  }
}

// ---------------- host orchestration ----------------
extern "C" void kernel_launch(void* const* d_in, const int* in_sizes, int n_in,
                              void* d_out, int out_size, void* d_ws, size_t ws_size,
                              hipStream_t stream) {
  const float* src = (const float*)d_in[0];
  const float* Wq  = (const float*)d_in[1];  const float* bq = (const float*)d_in[2];
  const float* Wk  = (const float*)d_in[3];  const float* bk = (const float*)d_in[4];
  const float* Wv  = (const float*)d_in[5];  const float* bv = (const float*)d_in[6];
  const float* Wo  = (const float*)d_in[7];  const float* bo = (const float*)d_in[8];
  const float* W1  = (const float*)d_in[9];  const float* b1 = (const float*)d_in[10];
  const float* W2  = (const float*)d_in[11]; const float* b2 = (const float*)d_in[12];
  const float* g1  = (const float*)d_in[13]; const float* be1 = (const float*)d_in[14];
  const float* g2  = (const float*)d_in[15]; const float* be2 = (const float*)d_in[16];

  char* ws = (char*)d_ws;
  const size_t WSZ = (size_t)D_MODEL * D_MODEL;        // one weight, elems
  bf16* WBF = (bf16*)ws;                               // 6 weights bf16 (12 MB)
  bf16* XN  = (bf16*)(ws + 12u * 1024 * 1024);         // 16 MB
  bf16* Qb  = (bf16*)(ws + 28u * 1024 * 1024);         // 16 MB (reused for FF1)
  bf16* Kb  = (bf16*)(ws + 44u * 1024 * 1024);         // 16 MB
  bf16* Vb  = (bf16*)(ws + 60u * 1024 * 1024);         // 16 MB
  bf16* AO  = (bf16*)(ws + 76u * 1024 * 1024);         // 16 MB
  float* X2 = (float*)(ws + 92u * 1024 * 1024);        // 32 MB
  float* OUT = (float*)d_out;

  // 1) weights -> bf16
  k_w2bf<<<dim3((D_MODEL * D_MODEL) / 256, 6), 256, 0, stream>>>(Wq, Wk, Wv, Wo, W1, W2, WBF);

  // 2) LN1
  k_layernorm<<<MTOT, 256, 0, stream>>>(src, g1, be1, XN);

  // 3) Q,K,V projections -> [B,H,S,HDIM] bf16
  const dim3 gg(MTOT / 32, D_MODEL / 256);
  k_gemm<<<gg, 256, 0, stream>>>(XN, WBF + 0 * WSZ, bq, nullptr, nullptr, Qb,
                                 MTOT, D_MODEL, D_MODEL, 4 | 8);
  k_gemm<<<gg, 256, 0, stream>>>(XN, WBF + 1 * WSZ, bk, nullptr, nullptr, Kb,
                                 MTOT, D_MODEL, D_MODEL, 4 | 8);
  k_gemm<<<gg, 256, 0, stream>>>(XN, WBF + 2 * WSZ, bv, nullptr, nullptr, Vb,
                                 MTOT, D_MODEL, D_MODEL, 4 | 8);

  // 4) attention
  k_attn<<<dim3(SEQ / 128, NHEADS, BATCH), 256, 0, stream>>>(Qb, Kb, Vb, AO);

  // 5) output projection + residual (x2 = src + attn @ Wo^T + bo)
  k_gemm<<<gg, 256, 0, stream>>>(AO, WBF + 3 * WSZ, bo, src, X2, nullptr,
                                 MTOT, D_MODEL, D_MODEL, 2);

  // 6) LN2
  k_layernorm<<<MTOT, 256, 0, stream>>>(X2, g2, be2, XN);

  // 7) FF1 with ReLU -> bf16 (reuse Q buffer)
  bf16* FF = Qb;
  k_gemm<<<gg, 256, 0, stream>>>(XN, WBF + 4 * WSZ, b1, nullptr, nullptr, FF,
                                 MTOT, D_MODEL, D_MODEL, 1 | 4);

  // 8) FF2 + residual -> final output fp32
  k_gemm<<<gg, 256, 0, stream>>>(FF, WBF + 5 * WSZ, b2, X2, OUT, nullptr,
                                 MTOT, D_MODEL, D_MODEL, 2);

  (void)in_sizes; (void)n_in; (void)out_size; (void)ws_size;
}
